// MoELayer_76888504533727
// MI455X (gfx1250) — matrix operand order repaired and checked by reading, not compiled, not run on hardware
//
#include <hip/hip_runtime.h>
#include <hip/hip_bf16.h>
#include <math.h>

#define T_TOKENS   2048
#define D_MODEL    1024
#define D_HIDDEN   4096
#define NUM_EXPERTS   8

typedef __attribute__((ext_vector_type(16))) __bf16       v16bf;
typedef __attribute__((ext_vector_type(8)))  float        v8f;
typedef __attribute__((ext_vector_type(4)))  unsigned int u32x4;
typedef __attribute__((ext_vector_type(8)))  int          i32x8;
typedef __attribute__((ext_vector_type(4)))  int          i32x4;

// LDS layout for the FFN kernel (dynamic shared memory, base offset 0)
#define X_STRIDE 1028                      // 1024 fp32 + 1 pad dword per 256 (TDM pad_enable)
#define H_STRIDE 4104                      // 4096 bf16 + 8 pad elems -> 2052 dword stride
#define LDS_X_BYTES (16 * X_STRIDE * 4)    // 65792
#define LDS_H_BYTES (16 * H_STRIDE * 2)    // 131328
#define LDS_TOTAL   (LDS_X_BYTES + LDS_H_BYTES + 128)

// Branchless GELU (sigmoid/tanh form): error ~1e-3, below bf16 quantization of h.
// v_exp_f32 + v_rcp_f32, no exec-mask divergence, no IEEE division sequence.
__device__ __forceinline__ float gelu_fast(float v) {
  float u = 1.5957691216f * v * __builtin_fmaf(0.044715f, v * v, 1.0f);
  return v * __builtin_amdgcn_rcpf(1.0f + __expf(-u));
}

// ---------------------------------------------------------------- zero output
__global__ void zero_kernel(float* __restrict__ p, int n) {
  int i = blockIdx.x * blockDim.x + threadIdx.x;
  if (i < n) p[i] = 0.0f;
}

// ---------------------------------------------------------------- gating + top2
__global__ __launch_bounds__(256) void gate_topk_kernel(
    const float* __restrict__ x, const float* __restrict__ gw, const float* __restrict__ gb,
    int* __restrict__ ti0, int* __restrict__ ti1,
    float* __restrict__ tw0, float* __restrict__ tw1)
{
  __shared__ float gwl[D_MODEL * NUM_EXPERTS];          // 32 KB staged gate weights
  for (int i = threadIdx.x; i < D_MODEL * NUM_EXPERTS; i += blockDim.x) gwl[i] = gw[i];
  __syncthreads();

  int t = blockIdx.x * blockDim.x + threadIdx.x;
  if (t >= T_TOKENS) return;

  float acc[NUM_EXPERTS];
  #pragma unroll
  for (int e = 0; e < NUM_EXPERTS; ++e) acc[e] = gb[e];
  const float* xr = x + (size_t)t * D_MODEL;
  for (int d = 0; d < D_MODEL; ++d) {
    float xv = xr[d];
    #pragma unroll
    for (int e = 0; e < NUM_EXPERTS; ++e) acc[e] += xv * gwl[d * NUM_EXPERTS + e];
  }
  // top-2 with first-occurrence tie-breaking (matches jax.lax.top_k)
  int i0 = 0; float l0 = acc[0];
  #pragma unroll
  for (int e = 1; e < NUM_EXPERTS; ++e) if (acc[e] > l0) { l0 = acc[e]; i0 = e; }
  int i1 = -1; float l1 = -3.4e38f;
  #pragma unroll
  for (int e = 0; e < NUM_EXPERTS; ++e) if (e != i0 && acc[e] > l1) { l1 = acc[e]; i1 = e; }
  float w0 = __builtin_amdgcn_rcpf(1.0f + __expf(l1 - l0));   // softmax over the two selected
  ti0[t] = i0; ti1[t] = i1; tw0[t] = w0; tw1[t] = 1.0f - w0;
}

// ------------------------------------------------- deterministic expert lists
__global__ __launch_bounds__(256) void build_lists_kernel(
    const int* __restrict__ ti0, const int* __restrict__ ti1,
    const float* __restrict__ tw0, const float* __restrict__ tw1,
    int* __restrict__ counts, int* __restrict__ ltok, float* __restrict__ lgate)
{
  int lane = threadIdx.x & 31;
  int e    = threadIdx.x >> 5;                          // 8 waves -> 8 experts
  int cnt = 0;
  for (int base = 0; base < T_TOKENS; base += 32) {
    int t = base + lane;
    bool m0 = (ti0[t] == e);
    bool m1 = (ti1[t] == e);
    unsigned bal = (unsigned)__ballot(m0 || m1);        // wave32: low 32 bits
    if (m0 || m1) {
      int pos = cnt + __popc(bal & ((1u << lane) - 1u));
      ltok [e * T_TOKENS + pos] = t;
      lgate[e * T_TOKENS + pos] = m0 ? tw0[t] : tw1[t];
    }
    cnt += __popc(bal);
  }
  if (lane == 0) counts[e] = cnt;
}

// ----------------------------------------------------------- WMMA frag loads
// A frag (16x32 bf16) from fp32 x-tile in LDS. Per ISA 7.12.2: lane half selects
// K blocks of 8; element e<8 -> K=k0+hb*8+e, e>=8 -> K=k0+16+hb*8+(e-8).
__device__ __forceinline__ v16bf load_a_x(const float* x_lds, int k0, int lane) {
  int m = lane & 15, hb = lane >> 4;
  int kA = k0 + hb * 8, kB = kA + 16;
  const float* rA = x_lds + m * X_STRIDE + kA + (kA >> 8);   // +1 dword pad per 256
  const float* rB = x_lds + m * X_STRIDE + kB + (kB >> 8);
  v16bf a;
  #pragma unroll
  for (int j = 0; j < 8; ++j) { a[j] = (__bf16)rA[j]; a[8 + j] = (__bf16)rB[j]; }
  return a;
}

// A frag from bf16 h-tile in LDS.
__device__ __forceinline__ v16bf load_a_h(const __bf16* h_lds, int k0, int lane) {
  int m = lane & 15, hb = lane >> 4;
  const __bf16* rA = h_lds + (size_t)m * H_STRIDE + k0 + hb * 8;
  v16bf a;
  #pragma unroll
  for (int j = 0; j < 8; ++j) { a[j] = rA[j]; a[8 + j] = rA[16 + j]; }
  return a;
}

// B frag (32x16) from row-major fp32 weights in global memory, same K order as A:
// element j<8 -> K=k0+hb*8+j, j>=8 -> K=k0+16+hb*8+(j-8);
// column n = lane&15.
__device__ __forceinline__ v16bf load_b_g(const float* base, int ld, int k0, int lane) {
  int n  = lane & 15;
  int kb = k0 + ((lane >> 4) << 3);
  v16bf b;
  #pragma unroll
  for (int j = 0; j < 16; ++j) b[j] = (__bf16)base[(size_t)(kb + j + ((j >> 3) << 3)) * ld + n];
  return b;
}

// ------------------------------------------------------------- fused expert FFN
__global__ __launch_bounds__(256) void moe_ffn_kernel(
    const float* __restrict__ x,
    const float* __restrict__ w1, const float* __restrict__ b1,
    const float* __restrict__ w2, const float* __restrict__ b2,
    const int* __restrict__ counts, const int* __restrict__ ltok,
    const float* __restrict__ lgate, float* __restrict__ out)
{
  int e    = blockIdx.x;
  int tile = blockIdx.y;
  int cnt  = counts[e];
  if (tile * 16 >= cnt) return;

  extern __shared__ char smem[];
  float*  x_lds  = (float*)smem;                              // offset 0 (TDM target)
  __bf16* h_lds  = (__bf16*)(smem + LDS_X_BYTES);
  int*    tok_s  = (int*)  (smem + LDS_X_BYTES + LDS_H_BYTES);
  float*  gate_s = (float*)(smem + LDS_X_BYTES + LDS_H_BYTES + 64);

  if (threadIdx.x < 16) {
    int idx = tile * 16 + (int)threadIdx.x;
    int src = (idx < cnt) ? idx : (cnt - 1);                  // pad rows: valid token, gate 0
    tok_s [threadIdx.x] = ltok [e * T_TOKENS + src];
    gate_s[threadIdx.x] = (idx < cnt) ? lgate[e * T_TOKENS + src] : 0.0f;
  }
  __syncthreads();

  const int lane = threadIdx.x & 31;
  const int wave = threadIdx.x >> 5;

  // ---- Stage 16 gathered token rows of x into LDS via TDM gather mode ----
#if __has_builtin(__builtin_amdgcn_tensor_load_to_lds)
  if (wave == 0) {
    unsigned long long ga = (unsigned long long)(const void*)x;
    u32x4 g0;
    g0[0] = 0x80000001u;                                      // gather_mode=1, 16b idx, count=1
    g0[1] = 0u;                                               // lds_addr: x_lds at offset 0
    g0[2] = (unsigned)(ga & 0xFFFFFFFFull);
    g0[3] = (unsigned)((ga >> 32) & 0x01FFFFFFull) | 0x80000000u;   // type=2 (image)
    i32x8 g1;
    g1[0] = (2 << 16) | (1 << 20) | (7 << 22);  // data_size=4B, pad_enable, pad 1dw / 256dw
    g1[1] = (int)((unsigned)D_MODEL  << 16);    // tensor_dim0[15:0] in bits 63:48
    g1[2] = (int)((unsigned)T_TOKENS << 16);    // tensor_dim1[15:0] in bits 95:80
    g1[3] = (int)((unsigned)D_MODEL  << 16);    // tile_dim0 = 1024 in bits 127:112
    g1[4] = 16;                                 // tile_dim1 = #valid indices
    g1[5] = D_MODEL;                            // tensor_dim0_stride low 32
    g1[6] = 0;
    g1[7] = 0;
    i32x4 g2, g3;                               // 16-bit row indices 0-7 / 8-15
    #pragma unroll
    for (int i = 0; i < 4; ++i) {
      g2[i] = (tok_s[2*i]     & 0xFFFF) | (tok_s[2*i + 1]     << 16);
      g3[i] = (tok_s[8 + 2*i] & 0xFFFF) | (tok_s[8 + 2*i + 1] << 16);
    }
    i32x8 gx = {0, 0, 0, 0, 0, 0, 0, 0};        // clang-23 6-arg form: extra group, zero
    __builtin_amdgcn_tensor_load_to_lds(g0, g1, g2, g3, gx, 0);
    __builtin_amdgcn_s_wait_tensorcnt(0);
  }
#else
  for (int i = threadIdx.x; i < 16 * D_MODEL; i += blockDim.x) {
    int m = i >> 10, k = i & (D_MODEL - 1);
    x_lds[m * X_STRIDE + k + (k >> 8)] = x[(size_t)tok_s[m] * D_MODEL + k];
  }
#endif
  __syncthreads();

  // ---- GEMM1: h[16,4096] = gelu(x_tile @ w1[e] + b1[e]), bf16 WMMA, f32 acc ----
  const float* w1e = w1 + (size_t)e * D_MODEL * D_HIDDEN;
  const float* b1e = b1 + (size_t)e * D_HIDDEN;
  for (int nt = wave; nt < D_HIDDEN / 16; nt += 8) {
    int n0 = nt * 16;
    v8f c = {};
    #pragma unroll 1
    for (int k0 = 0; k0 < D_MODEL; k0 += 32) {
      v16bf a = load_a_x(x_lds, k0, lane);
      v16bf b = load_b_g(w1e + n0, D_HIDDEN, k0, lane);
      c = __builtin_amdgcn_wmma_f32_16x16x32_bf16(false, a, false, b, (short)0, c, false, false);
    }
    int col = n0 + (lane & 15);
    int mb  = (lane < 16) ? 0 : 8;                            // C layout: vgpr r -> row r / r+8
    float bias = b1e[col];
    #pragma unroll
    for (int r = 0; r < 8; ++r) {
      h_lds[(size_t)(mb + r) * H_STRIDE + col] = (__bf16)gelu_fast(c[r] + bias);
    }
  }
  __syncthreads();

  // ---- GEMM2: y[16,1024] = h @ w2[e] + b2[e]; gate-scale; atomic combine ----
  const float* w2e = w2 + (size_t)e * D_HIDDEN * D_MODEL;
  const float* b2e = b2 + (size_t)e * D_MODEL;
  for (int nt = wave; nt < D_MODEL / 16; nt += 8) {
    int n0 = nt * 16;
    v8f c = {};
    #pragma unroll 1
    for (int k0 = 0; k0 < D_HIDDEN; k0 += 32) {
      v16bf a = load_a_h(h_lds, k0, lane);
      v16bf b = load_b_g(w2e + n0, D_MODEL, k0, lane);
      c = __builtin_amdgcn_wmma_f32_16x16x32_bf16(false, a, false, b, (short)0, c, false, false);
    }
    int col = n0 + (lane & 15);
    int mb  = (lane < 16) ? 0 : 8;
    float bias = b2e[col];
    #pragma unroll
    for (int r = 0; r < 8; ++r) {
      int m = mb + r;
      float v = (c[r] + bias) * gate_s[m];
      atomicAdd(out + (size_t)tok_s[m] * D_MODEL + col, v);   // exactly 2 adds/elem: deterministic
    }
  }
}

// ---------------------------------------------------------------- launcher
extern "C" void kernel_launch(void* const* d_in, const int* in_sizes, int n_in,
                              void* d_out, int out_size, void* d_ws, size_t ws_size,
                              hipStream_t stream)
{
  const float* x  = (const float*)d_in[0];
  const float* gw = (const float*)d_in[1];
  const float* gb = (const float*)d_in[2];
  const float* w1 = (const float*)d_in[3];
  const float* b1 = (const float*)d_in[4];
  const float* w2 = (const float*)d_in[5];
  const float* b2 = (const float*)d_in[6];
  float* out = (float*)d_out;

  char* ws = (char*)d_ws;
  int*   ti0    = (int*)ws;    ws += T_TOKENS * 4;
  int*   ti1    = (int*)ws;    ws += T_TOKENS * 4;
  float* tw0    = (float*)ws;  ws += T_TOKENS * 4;
  float* tw1    = (float*)ws;  ws += T_TOKENS * 4;
  int*   counts = (int*)ws;    ws += 32;
  int*   ltok   = (int*)ws;    ws += NUM_EXPERTS * T_TOKENS * 4;
  float* lgate  = (float*)ws;  /* ws += NUM_EXPERTS * T_TOKENS * 4; */

  int n_out = T_TOKENS * D_MODEL;
  zero_kernel<<<(n_out + 255) / 256, 256, 0, stream>>>(out, n_out);
  gate_topk_kernel<<<T_TOKENS / 256, 256, 0, stream>>>(x, gw, gb, ti0, ti1, tw0, tw1);
  build_lists_kernel<<<1, 256, 0, stream>>>(ti0, ti1, tw0, tw1, counts, ltok, lgate);
  dim3 grid(NUM_EXPERTS, T_TOKENS / 16);        // max 128 tiles/expert; extras early-exit
  moe_ffn_kernel<<<grid, 256, LDS_TOTAL, stream>>>(x, w1, b1, w2, b2,
                                                   counts, ltok, lgate, out);
}
